// MT_CNN_LSTM_B_36438502539673
// MI455X (gfx1250) — matrix operand order repaired and checked by reading, not compiled, not run on hardware
//
#include <hip/hip_runtime.h>
#include <cstdint>
#include <cstddef>

typedef __bf16 bf16_t;
typedef __attribute__((ext_vector_type(16))) __bf16 v16bf;
typedef __attribute__((ext_vector_type(8)))  float  v8f;

// Async direct global->LDS path (CDNA5 GLOBAL_LOAD_ASYNC_TO_LDS_*, ASYNCcnt).
// Guarded: falls back to load+ds_store pipeline if builtins are unavailable.
#if __has_builtin(__builtin_amdgcn_global_load_async_to_lds_b128) && \
    __has_builtin(__builtin_amdgcn_s_wait_asynccnt)
#define ASYNC_LDS 1
#else
#define ASYNC_LDS 0
#endif

// ---------------------------------------------------------------------------
// WMMA fragment loaders (layouts per CDNA5 ISA §7.12.2, wave32)
// ---------------------------------------------------------------------------
union FragU { v16bf v; uint4 q[2]; };

// A matrix 16x32 bf16, row-major source with leading dim ld (elements).
static __device__ __forceinline__ v16bf frag_a_load(const bf16_t* __restrict__ tile, int ld) {
  const int lane = threadIdx.x & 31;
  const int m  = lane & 15;
  const int kb = (lane & 16) ? 8 : 0;
  const bf16_t* p = tile + (size_t)m * ld + kb;
  FragU u;
  u.q[0] = *reinterpret_cast<const uint4*>(p);
  u.q[1] = *reinterpret_cast<const uint4*>(p + 16);
  return u.v;
}

// B matrix 32x16 bf16 from an LDS slab with row stride LS elements.
template <int LS>
static __device__ __forceinline__ v16bf frag_b_lds(const bf16_t* __restrict__ slab) {
  const int lane = threadIdx.x & 31;
  const int n  = lane & 15;
  const int kb = (lane & 16) ? 8 : 0;
  const bf16_t* p = slab + (size_t)n * LS + kb;
  FragU u;
  u.q[0] = *reinterpret_cast<const uint4*>(p);   // ds_load_b128
  u.q[1] = *reinterpret_cast<const uint4*>(p + 16);
  return u.v;
}

static __device__ __forceinline__ float sigf(float x) {
  return 1.0f / (1.0f + __expf(-x));
}

// ---------------------------------------------------------------------------
// Prep kernels
// ---------------------------------------------------------------------------
__global__ void k_f32_to_bf16(const float* __restrict__ s, bf16_t* __restrict__ d, int n) {
  int i = blockIdx.x * blockDim.x + threadIdx.x;
  if (i < n) d[i] = (bf16_t)s[i];
}

__global__ void k_bias_sum(const float* __restrict__ a, const float* __restrict__ b,
                           float* __restrict__ o, int n) {
  int i = blockIdx.x * blockDim.x + threadIdx.x;
  if (i < n) o[i] = a[i] + b[i];
}

__global__ void k_zero_u32(uint32_t* __restrict__ p, int n) {
  int i = blockIdx.x * blockDim.x + threadIdx.x;
  if (i < n) p[i] = 0u;
}

// ---------------------------------------------------------------------------
// Conv branch kernels (faithful to the reference's view-scrambles)
// ---------------------------------------------------------------------------
__global__ void k_conv_ts_m3(const float* __restrict__ x, const float* __restrict__ w,
                             const float* __restrict__ bias, float* __restrict__ out) {
  int tid = blockIdx.x * blockDim.x + threadIdx.x;        // 1024*64*64 threads
  int f  = tid & 63;
  int s2 = (tid >> 6) & 63;
  int b  = tid >> 12;
  float v[6];
#pragma unroll
  for (int d = 0; d < 6; ++d) {
    int s = 2 * s2 - 2 + d;
    v[d] = (s >= 0 && s < 128) ? x[(size_t)b * 8192 + (size_t)s * 64 + f] : 0.0f;
  }
  float sum = 0.0f;
#pragma unroll
  for (int k = 0; k < 16; ++k) {
    float a0 = bias[k], a1 = bias[k];
#pragma unroll
    for (int d = 0; d < 5; ++d) {
      float wk = w[k * 5 + d];
      a0 += wk * v[d];
      a1 += wk * v[d + 1];
    }
    sum += fmaxf(a0, 0.0f) + fmaxf(a1, 0.0f);
  }
  out[(size_t)b * 4096 + (size_t)s2 * 64 + f] = sum * (1.0f / 32.0f);
}

__global__ void k_conv_f_m3(const float* __restrict__ cts, const float* __restrict__ w,
                            const float* __restrict__ bias, bf16_t* __restrict__ x0) {
  int tid = blockIdx.x * blockDim.x + threadIdx.x;        // 65536*64 threads
  int fo = tid & 63;
  int u  = tid >> 6;
  int sp = u & 63;
  int bp = u >> 6;
  size_t j = (size_t)sp * 1024 + bp;
  const float* row = cts + j * 64;
  float vm = (fo > 0)  ? row[fo - 1] : 0.0f;
  float vc = row[fo];
  float vp = (fo < 63) ? row[fo + 1] : 0.0f;
  float sum = 0.0f;
#pragma unroll
  for (int k = 0; k < 16; ++k) {
    float a = bias[k] + w[k * 3 + 0] * vm + w[k * 3 + 1] * vc + w[k * 3 + 2] * vp;
    sum += fmaxf(a, 0.0f);
  }
  x0[(size_t)u * 64 + fo] = (bf16_t)(sum * (1.0f / 16.0f));
}

__global__ void k_conv_f_m30(const float* __restrict__ xm, const float* __restrict__ w,
                             const float* __restrict__ bias, bf16_t* __restrict__ x0) {
  int tid = blockIdx.x * blockDim.x + threadIdx.x;        // 65536*32 threads
  int fo = tid & 31;
  int u  = tid >> 5;
  int sp = u & 63;
  int bp = u >> 6;
  size_t j = (size_t)sp * 1024 + bp;
  const float* row = xm + j * 32;
  float vm = (fo > 0)  ? row[fo - 1] : 0.0f;
  float vc = row[fo];
  float vp = (fo < 31) ? row[fo + 1] : 0.0f;
  float sum = 0.0f;
#pragma unroll
  for (int k = 0; k < 16; ++k) {
    float a = bias[k] + w[k * 3 + 0] * vm + w[k * 3 + 1] * vc + w[k * 3 + 2] * vp;
    sum += fmaxf(a, 0.0f);
  }
  x0[(size_t)u * 32 + fo] = (bf16_t)(sum * (1.0f / 16.0f));
}

// ---------------------------------------------------------------------------
// LSTM step building blocks.
// Block = 256 thr (8 waves). WG tile: 128 batch rows x 64 h-cols x 4 gates.
// Wave (4 M x 2 Hcol): 32 rows x 32 hcols x 4 gates = 16 accumulators (v8f).
// Weight slab (256 W-rows x 32 K) double-buffered in LDS, stride 40 bf16
// (80B keeps 16B alignment; 20*n mod 64 is a permutation -> conflict-free).
// Staging uses GLOBAL_LOAD_ASYNC_TO_LDS_B128 (ASYNCcnt) when available.
// ---------------------------------------------------------------------------
static constexpr int KC  = 32;                   // K per slab (one WMMA k-step)
static constexpr int LS  = 40;                   // LDS row stride (elements)
static constexpr int SLAB = 256 * LS;            // one buffer (elements)
static constexpr int LDS_ELEMS = 2 * SLAB;       // double buffer: 40KB

// Copy KC bf16 (64B) from global row to LDS row.
static __device__ __forceinline__ void stage_row(const bf16_t* __restrict__ gsrc,
                                                 bf16_t* __restrict__ ldst) {
#if ASYNC_LDS
  typedef int v4i __attribute__((vector_size(16)));
  typedef __attribute__((address_space(1))) v4i gv4i;
  typedef __attribute__((address_space(3))) v4i lv4i;
  gv4i* g = (gv4i*)(bf16_t*)gsrc;   // drop const, generic -> AS1, retype to v4i
  lv4i* l = (lv4i*)ldst;            // generic -> AS3, retype to v4i
#pragma unroll
  for (int q = 0; q < 4; ++q)
    __builtin_amdgcn_global_load_async_to_lds_b128(g + q, l + q, 0, 0);
#else
  const uint4* s = reinterpret_cast<const uint4*>(gsrc);
  uint4* d = reinterpret_cast<uint4*>(ldst);
#pragma unroll
  for (int q = 0; q < 4; ++q) d[q] = s[q];
#endif
}

static __device__ __forceinline__ void wait_async_all() {
#if ASYNC_LDS
  __builtin_amdgcn_s_wait_asynccnt(0);
#endif
}

template <int KTOT, int H>
static __device__ __forceinline__ void gemm_slabs(
    const bf16_t* __restrict__ A,     // (1024, KTOT) row-major
    const bf16_t* __restrict__ W,     // (4H, KTOT) row-major
    bf16_t* __restrict__ sB,          // LDS, 2*SLAB elements
    int row0, int hwg, int whc,
    v8f acc[2][4][2]) {
  constexpr int NK = KTOT / KC;
  const int t = threadIdx.x;
  const int sg = t >> 6;          // gate group 0..3 for staging
  const int sc = t & 63;          // h-col within group for staging
  const bf16_t* wrow = W + (size_t)(sg * H + hwg + sc) * KTOT;
  bf16_t* lrow = sB + (size_t)(sg * 64 + sc) * LS;

  stage_row(wrow, lrow);          // prologue: slab 0
#pragma unroll
  for (int i = 0; i < NK; ++i) {
    wait_async_all();
    __syncthreads();              // slab i ready & visible to all waves
    if (i + 1 < NK)               // prefetch next slab into other buffer
      stage_row(wrow + (i + 1) * KC, lrow + ((i + 1) & 1) * SLAB);

    const bf16_t* cur = sB + (i & 1) * SLAB;
    v16bf a0 = frag_a_load(A + (size_t)row0 * KTOT + i * KC, KTOT);
    v16bf a1 = frag_a_load(A + (size_t)(row0 + 16) * KTOT + i * KC, KTOT);
#pragma unroll
    for (int g = 0; g < 4; ++g)
#pragma unroll
      for (int j = 0; j < 2; ++j) {
        v16bf b = frag_b_lds<LS>(cur + (size_t)(g * 64 + whc + j * 16) * LS);
        acc[0][g][j] = __builtin_amdgcn_wmma_f32_16x16x32_bf16(
            false, a0, false, b, (short)0, acc[0][g][j], false, false);
        acc[1][g][j] = __builtin_amdgcn_wmma_f32_16x16x32_bf16(
            false, a1, false, b, (short)0, acc[1][g][j], false, false);
      }
    __syncthreads();              // all waves done reading slab i
  }
}

template <int I, int H>
static __device__ __forceinline__ void lstm_body(
    const bf16_t* __restrict__ x,      // (B, I) bf16
    const bf16_t* __restrict__ wih,    // (4H, I) bf16
    const bf16_t* __restrict__ whh,    // (4H, H) bf16
    const float*  __restrict__ bsum,   // (4H,) f32
    const bf16_t* __restrict__ hprev,  // (B, H) bf16 (ping)
    bf16_t*       __restrict__ hout,   // (B, H) bf16 (pong)
    float*        __restrict__ houtf,  // (B, H) f32
    float*        __restrict__ c,      // (B, H) f32 in-place
    bf16_t* __restrict__ sB) {
  const int wave  = threadIdx.x >> 5;
  const int waveM = wave >> 1;                       // 0..3
  const int waveH = wave & 1;                        // 0..1
  const int row0  = blockIdx.x * 128 + waveM * 32;   // batch rows (32 per wave)
  const int hwg   = blockIdx.y * 64;                 // WG h-col base
  const int whc   = waveH * 32;                      // wave h-col offset

  v8f acc[2][4][2];
#pragma unroll
  for (int mi = 0; mi < 2; ++mi)
#pragma unroll
    for (int g = 0; g < 4; ++g)
#pragma unroll
      for (int j = 0; j < 2; ++j)
#pragma unroll
        for (int e = 0; e < 8; ++e) acc[mi][g][j][e] = 0.0f;

  gemm_slabs<I, H>(x,     wih, sB, row0, hwg, whc, acc);
  gemm_slabs<H, H>(hprev, whh, sB, row0, hwg, whc, acc);

  // Pointwise gates. C/D: elem e -> row = base + e + (lane>=16?8:0), col = lane&15.
  const int lane = threadIdx.x & 31;
  const int nlo  = lane & 15;
  const int mhi  = (lane & 16) ? 8 : 0;
#pragma unroll
  for (int j = 0; j < 2; ++j) {
    const int col = hwg + whc + j * 16 + nlo;
    const float bi = bsum[0 * H + col];
    const float bf = bsum[1 * H + col];
    const float bg = bsum[2 * H + col];
    const float bo = bsum[3 * H + col];
#pragma unroll
    for (int mi = 0; mi < 2; ++mi)
#pragma unroll
      for (int e = 0; e < 8; ++e) {
        const int row = row0 + mi * 16 + mhi + e;
        const size_t idx = (size_t)row * H + col;
        float zi = acc[mi][0][j][e] + bi;
        float zf = acc[mi][1][j][e] + bf;
        float zg = acc[mi][2][j][e] + bg;
        float zo = acc[mi][3][j][e] + bo;
        float cp = c[idx];
        float cn = sigf(zf) * cp + sigf(zi) * tanhf(zg);
        float hn = sigf(zo) * tanhf(cn);
        c[idx]     = cn;
        houtf[idx] = hn;
        hout[idx]  = (bf16_t)hn;
      }
  }
}

// Fused both-branch timestep kernel: blockIdx.z==0 -> m3, ==1 -> m30.
template <int I3, int H3, int I30, int H30>
__global__ __launch_bounds__(256)
void k_lstm_step2(const bf16_t* __restrict__ x3,  const bf16_t* __restrict__ wih3,
                  const bf16_t* __restrict__ whh3, const float* __restrict__ bs3,
                  const bf16_t* __restrict__ hp3,  bf16_t* __restrict__ ho3,
                  float* __restrict__ hof3,        float* __restrict__ c3,
                  const bf16_t* __restrict__ x30,  const bf16_t* __restrict__ wih30,
                  const bf16_t* __restrict__ whh30,const float* __restrict__ bs30,
                  const bf16_t* __restrict__ hp30, bf16_t* __restrict__ ho30,
                  float* __restrict__ hof30,       float* __restrict__ c30) {
  __shared__ bf16_t sB[LDS_ELEMS];
  if (blockIdx.z == 0) {
    lstm_body<I3, H3>(x3, wih3, whh3, bs3, hp3, ho3, hof3, c3, sB);
  } else {
    if (blockIdx.y >= H30 / 64) return;
    lstm_body<I30, H30>(x30, wih30, whh30, bs30, hp30, ho30, hof30, c30, sB);
  }
}

// ---------------------------------------------------------------------------
// FC head (tiny)
// ---------------------------------------------------------------------------
__global__ void k_fc_cat_relu(const float* __restrict__ a, int Ka,
                              const float* __restrict__ b, int Kb,
                              const float* __restrict__ W, const float* __restrict__ bias,
                              float* __restrict__ out, int N, int Bn) {
  int t = blockIdx.x * blockDim.x + threadIdx.x;
  if (t >= Bn * N) return;
  int o = t % N, r = t / N;
  const float* w = W + (size_t)o * (Ka + Kb);
  float s = bias[o];
  for (int k = 0; k < Ka; ++k) s += a[(size_t)r * Ka + k] * w[k];
  for (int k = 0; k < Kb; ++k) s += b[(size_t)r * Kb + k] * w[Ka + k];
  out[t] = fmaxf(s, 0.0f);
}

__global__ void k_fc(const float* __restrict__ in, int K,
                     const float* __restrict__ W, const float* __restrict__ bias,
                     float* __restrict__ out, int N, int Bn) {
  int t = blockIdx.x * blockDim.x + threadIdx.x;
  if (t >= Bn * N) return;
  int o = t % N, r = t / N;
  const float* w = W + (size_t)o * K;
  float s = bias[o];
  for (int k = 0; k < K; ++k) s += in[(size_t)r * K + k] * w[k];
  out[t] = s;
}

// ---------------------------------------------------------------------------
extern "C" void kernel_launch(void* const* d_in, const int* in_sizes, int n_in,
                              void* d_out, int out_size, void* d_ws, size_t ws_size,
                              hipStream_t stream) {
  (void)in_sizes; (void)n_in; (void)out_size; (void)ws_size;
  const float* x_m3      = (const float*)d_in[0];
  const float* x_m30     = (const float*)d_in[1];
  const float* w_ts      = (const float*)d_in[2];
  const float* b_ts      = (const float*)d_in[3];
  const float* w_f_m3    = (const float*)d_in[4];
  const float* b_f_m3    = (const float*)d_in[5];
  const float* w_ih0_m3  = (const float*)d_in[6];
  const float* w_hh0_m3  = (const float*)d_in[7];
  const float* b_ih0_m3  = (const float*)d_in[8];
  const float* b_hh0_m3  = (const float*)d_in[9];
  const float* w_ih1_m3  = (const float*)d_in[10];
  const float* w_hh1_m3  = (const float*)d_in[11];
  const float* b_ih1_m3  = (const float*)d_in[12];
  const float* b_hh1_m3  = (const float*)d_in[13];
  const float* w_f_m30   = (const float*)d_in[14];
  const float* b_f_m30   = (const float*)d_in[15];
  const float* w_ih0_m30 = (const float*)d_in[16];
  const float* w_hh0_m30 = (const float*)d_in[17];
  const float* b_ih0_m30 = (const float*)d_in[18];
  const float* b_hh0_m30 = (const float*)d_in[19];
  const float* w_ih1_m30 = (const float*)d_in[20];
  const float* w_hh1_m30 = (const float*)d_in[21];
  const float* b_ih1_m30 = (const float*)d_in[22];
  const float* b_hh1_m30 = (const float*)d_in[23];
  const float* w_fc1     = (const float*)d_in[24];
  const float* b_fc1     = (const float*)d_in[25];
  const float* w_fc2     = (const float*)d_in[26];
  const float* b_fc2     = (const float*)d_in[27];
  const float* w_fc1_30  = (const float*)d_in[28];
  const float* b_fc1_30  = (const float*)d_in[29];
  const float* w_fc2_30  = (const float*)d_in[30];
  const float* b_fc2_30  = (const float*)d_in[31];

  // --- workspace bump allocator ---
  char* base = (char*)d_ws;
  size_t off = 0;
  auto alloc = [&](size_t bytes) -> char* {
    char* p = base + off;
    off = (off + bytes + 255) & ~(size_t)255;
    return p;
  };
  float*  cts     = (float*)alloc((size_t)1024 * 64 * 64 * 4);
  bf16_t* x0m3    = (bf16_t*)alloc((size_t)64 * 1024 * 64 * 2);
  bf16_t* x0m30   = (bf16_t*)alloc((size_t)64 * 1024 * 32 * 2);
  bf16_t* bwih0m3 = (bf16_t*)alloc((size_t)1024 * 64 * 2);
  bf16_t* bwhh0m3 = (bf16_t*)alloc((size_t)1024 * 256 * 2);
  bf16_t* bwih1m3 = (bf16_t*)alloc((size_t)1024 * 256 * 2);
  bf16_t* bwhh1m3 = (bf16_t*)alloc((size_t)1024 * 256 * 2);
  bf16_t* bwih0m30= (bf16_t*)alloc((size_t)512 * 32 * 2);
  bf16_t* bwhh0m30= (bf16_t*)alloc((size_t)512 * 128 * 2);
  bf16_t* bwih1m30= (bf16_t*)alloc((size_t)512 * 128 * 2);
  bf16_t* bwhh1m30= (bf16_t*)alloc((size_t)512 * 128 * 2);
  float*  bs0m3   = (float*)alloc(1024 * 4);
  float*  bs1m3   = (float*)alloc(1024 * 4);
  float*  bs0m30  = (float*)alloc(512 * 4);
  float*  bs1m30  = (float*)alloc(512 * 4);
  bf16_t* h0m3    = (bf16_t*)alloc((size_t)2 * 1024 * 256 * 2);  // ping-pong
  bf16_t* h1m3    = (bf16_t*)alloc((size_t)2 * 1024 * 256 * 2);
  bf16_t* h0m30   = (bf16_t*)alloc((size_t)2 * 1024 * 128 * 2);
  bf16_t* h1m30   = (bf16_t*)alloc((size_t)2 * 1024 * 128 * 2);
  float*  c0m3    = (float*)alloc((size_t)1024 * 256 * 4);
  float*  c1m3    = (float*)alloc((size_t)1024 * 256 * 4);
  float*  c0m30   = (float*)alloc((size_t)1024 * 128 * 4);
  float*  c1m30   = (float*)alloc((size_t)1024 * 128 * 4);
  float*  hf0m3   = (float*)alloc((size_t)1024 * 256 * 4);
  float*  hf1m3   = (float*)alloc((size_t)1024 * 256 * 4);
  float*  hf0m30  = (float*)alloc((size_t)1024 * 128 * 4);
  float*  hf1m30  = (float*)alloc((size_t)1024 * 128 * 4);
  float*  tmp1    = (float*)alloc((size_t)1024 * 192 * 4);
  float*  tmp2    = (float*)alloc((size_t)1024 * 64 * 4);

  // --- weight prep ---
  auto cvt = [&](const float* s, bf16_t* d, int n) {
    k_f32_to_bf16<<<(n + 255) / 256, 256, 0, stream>>>(s, d, n);
  };
  cvt(w_ih0_m3,  bwih0m3,  1024 * 64);
  cvt(w_hh0_m3,  bwhh0m3,  1024 * 256);
  cvt(w_ih1_m3,  bwih1m3,  1024 * 256);
  cvt(w_hh1_m3,  bwhh1m3,  1024 * 256);
  cvt(w_ih0_m30, bwih0m30, 512 * 32);
  cvt(w_hh0_m30, bwhh0m30, 512 * 128);
  cvt(w_ih1_m30, bwih1m30, 512 * 128);
  cvt(w_hh1_m30, bwhh1m30, 512 * 128);
  k_bias_sum<<<4, 256, 0, stream>>>(b_ih0_m3, b_hh0_m3, bs0m3, 1024);
  k_bias_sum<<<4, 256, 0, stream>>>(b_ih1_m3, b_hh1_m3, bs1m3, 1024);
  k_bias_sum<<<2, 256, 0, stream>>>(b_ih0_m30, b_hh0_m30, bs0m30, 512);
  k_bias_sum<<<2, 256, 0, stream>>>(b_ih1_m30, b_hh1_m30, bs1m30, 512);

  // --- zero-init recurrent state (ws is poisoned) ---
  auto zero = [&](void* p, size_t bytes) {
    int n = (int)(bytes / 4);
    k_zero_u32<<<(n + 255) / 256, 256, 0, stream>>>((uint32_t*)p, n);
  };
  zero(h0m3,  (size_t)2 * 1024 * 256 * 2);
  zero(h1m3,  (size_t)2 * 1024 * 256 * 2);
  zero(h0m30, (size_t)2 * 1024 * 128 * 2);
  zero(h1m30, (size_t)2 * 1024 * 128 * 2);
  zero(c0m3,  (size_t)1024 * 256 * 4);
  zero(c1m3,  (size_t)1024 * 256 * 4);
  zero(c0m30, (size_t)1024 * 128 * 4);
  zero(c1m30, (size_t)1024 * 128 * 4);

  // --- conv front-ends ---
  k_conv_ts_m3<<<(1024 * 64 * 64) / 256, 256, 0, stream>>>(x_m3, w_ts, b_ts, cts);
  k_conv_f_m3 <<<(65536 * 64) / 256, 256, 0, stream>>>(cts, w_f_m3, b_f_m3, x0m3);
  k_conv_f_m30<<<(65536 * 32) / 256, 256, 0, stream>>>(x_m30, w_f_m30, b_f_m30, x0m30);

  // --- LSTM recurrence: 64 steps, layer0 (both branches) then layer1 ---
  const size_t s3 = (size_t)1024 * 256, s30 = (size_t)1024 * 128;
  for (int t = 0; t < 64; ++t) {
    const int pi = t & 1, po = (t + 1) & 1;
    // layer 0: m3 (I=64,H=256) + m30 (I=32,H=128)
    k_lstm_step2<64, 256, 32, 128><<<dim3(8, 4, 2), 256, 0, stream>>>(
        x0m3 + (size_t)t * 1024 * 64, bwih0m3, bwhh0m3, bs0m3,
        h0m3 + pi * s3, h0m3 + po * s3, hf0m3, c0m3,
        x0m30 + (size_t)t * 1024 * 32, bwih0m30, bwhh0m30, bs0m30,
        h0m30 + pi * s30, h0m30 + po * s30, hf0m30, c0m30);
    // layer 1: m3 (I=256,H=256) + m30 (I=128,H=128); x = layer0's fresh h
    k_lstm_step2<256, 256, 128, 128><<<dim3(8, 4, 2), 256, 0, stream>>>(
        h0m3 + po * s3, bwih1m3, bwhh1m3, bs1m3,
        h1m3 + pi * s3, h1m3 + po * s3, hf1m3, c1m3,
        h0m30 + po * s30, bwih1m30, bwhh1m30, bs1m30,
        h1m30 + pi * s30, h1m30 + po * s30, hf1m30, c1m30);
  }

  // --- FC heads ---
  float* yout  = (float*)d_out;            // (1024, 24)
  float* yout2 = yout + 1024 * 24;         // (1024, 8)
  k_fc_cat_relu<<<(1024 * 192 + 255) / 256, 256, 0, stream>>>(
      hf1m3, 256, hf1m30, 128, w_fc1, b_fc1, tmp1, 192, 1024);
  k_fc<<<(1024 * 24 + 255) / 256, 256, 0, stream>>>(
      tmp1, 192, w_fc2, b_fc2, yout, 24, 1024);
  k_fc_cat_relu<<<(1024 * 64 + 255) / 256, 256, 0, stream>>>(
      hf1m30, 128, hf1m30, 0, w_fc1_30, b_fc1_30, tmp2, 64, 1024);
  k_fc<<<(1024 * 8 + 255) / 256, 256, 0, stream>>>(
      tmp2, 64, w_fc2_30, b_fc2_30, yout2, 8, 1024);
}
